// LocalSelfAttention_15693810499609
// MI455X (gfx1250) — compile-verified, matrix operand order checked by reading, not run on hardware
//
#include <hip/hip_runtime.h>
#include <hip/hip_bf16.h>
#include <math.h>
#include <stdint.h>

// ---------------------------------------------------------------------------
// LocalSelfAttention for MI455X (gfx1250, wave32, WMMA + TDM)
//   DIM=768, LOCAL_DIM=20, D=12,H=24,W=24 -> N=6912, B=2, BN=13824
//   window per axis: d in [-1,1], h in [-3,2], w in [-3,2]  (<=108 keys)
// ---------------------------------------------------------------------------

#define DIMC   768
#define LD     20
#define DD     12
#define HH     24
#define WW     24
#define NTOK   (DD*HH*WW)      // 6912
#define BB     2
#define BN     (BB*NTOK)       // 13824
#define QKVP   64              // padded 60 -> 64 cols for qkv
#define OPAD   32              // padded 20 -> 32 cols for attention output
#define APAD   776             // LDS halves per row in k1 (768 + 8 pad, 16B aligned)

typedef __attribute__((ext_vector_type(16))) _Float16 v16h;
typedef __attribute__((ext_vector_type(8)))  _Float16 v8h;
typedef __attribute__((ext_vector_type(8)))  float    v8f;
typedef __attribute__((ext_vector_type(4)))  unsigned int v4u;
typedef __attribute__((ext_vector_type(8)))  int          v8i;
typedef __attribute__((ext_vector_type(4)))  int          v4i;

union V16 { v16h v; v8h h[2]; };

#ifndef __has_builtin
#define __has_builtin(x) 0
#endif
#if __has_builtin(__builtin_amdgcn_tensor_load_to_lds)
#define HAVE_TDM 1
#else
#define HAVE_TDM 0
#endif

// ---------------------------------------------------------------------------
// k0: prep weights in workspace.
//   WcT  f16 [64][768]   : WcT[n][k] = sum_l Wd[k][l]*Wqkv[l][n]   (n<60, else 0)
//   bc   f32 [64]        : bc[n]     = sum_l bd[l]*Wqkv[l][n]
//   WuT  f16 [768][32]   : WuT[c][l] = Wu[l][c] for l<20 else 0
// ---------------------------------------------------------------------------
__global__ __launch_bounds__(64) void k0_prep(const float* __restrict__ Wd,
                                              const float* __restrict__ bd,
                                              const float* __restrict__ Wqkv,
                                              const float* __restrict__ Wu,
                                              _Float16* __restrict__ WcT,
                                              float* __restrict__ bc,
                                              _Float16* __restrict__ WuT) {
  int blk = blockIdx.x;
  if (blk < QKVP) {
    int n = blk;
    for (int k = threadIdx.x; k < DIMC; k += 64) {
      float s = 0.f;
      if (n < 3*LD) {
        #pragma unroll
        for (int l = 0; l < LD; ++l) s += Wd[k*LD + l] * Wqkv[l*(3*LD) + n];
      }
      WcT[n*DIMC + k] = (_Float16)s;
    }
    if (threadIdx.x == 0) {
      float s = 0.f;
      if (n < 3*LD) {
        #pragma unroll
        for (int l = 0; l < LD; ++l) s += bd[l] * Wqkv[l*(3*LD) + n];
      }
      bc[n] = s;
    }
  } else {
    int c = blk - QKVP;           // 0..767
    int l = threadIdx.x;
    if (l < OPAD)
      WuT[c*OPAD + l] = (l < LD) ? (_Float16)Wu[l*DIMC + c] : (_Float16)0.f;
  }
}

// ---------------------------------------------------------------------------
// k1: fused LayerNorm + qkv projection (xn @ Wc + bc) via WMMA f16.
//   grid = BN/16 blocks, 128 threads (4 waves).
//   x tile staged into LDS by the Tensor Data Mover (one 4x768 f32 slice per
//   wave -> four concurrent TDM streams; each wave waits on its own TENSORcnt
//   and LayerNorms exactly the rows it DMA'd).
//   qkv f32 [BN][64] (cols 0..59 valid: q|k|v each 20)
// ---------------------------------------------------------------------------
__global__ __launch_bounds__(128) void k1_ln_qkv(const float* __restrict__ x,
                                                 const float* __restrict__ gamma,
                                                 const float* __restrict__ beta,
                                                 const _Float16* __restrict__ WcT,
                                                 const float* __restrict__ bc,
                                                 float* __restrict__ qkv) {
  __shared__ __attribute__((aligned(16))) float    Xf[16 * DIMC];   // 48 KB
  __shared__ __attribute__((aligned(16))) _Float16 Ash[16 * APAD];  // 24 KB
  const int row0 = blockIdx.x * 16;
  const int lane = threadIdx.x & 31;
  const int wave = __builtin_amdgcn_readfirstlane(threadIdx.x >> 5);

#if HAVE_TDM
  {
    // Tensor DMA: 4 rows x 768 f32 (12 KB), rows row0+4*wave .. +3
    const unsigned lds_addr =
        (unsigned)(uintptr_t)&Xf[0] + (unsigned)(wave * 4 * DIMC * 4);
    const unsigned long long ga =
        (unsigned long long)(uintptr_t)(x + (size_t)(row0 + wave * 4) * DIMC);
    v4u g0;
    g0[0] = 1u;                                   // count=1 (valid user D#)
    g0[1] = lds_addr;                             // LDS byte address
    g0[2] = (unsigned)(ga & 0xffffffffu);         // global_addr[31:0]
    g0[3] = (unsigned)((ga >> 32) & 0x01ffffffu)  // global_addr[56:32]
            | (2u << 30);                         // type=2 ("image")
    v8i g1;
    g1[0] = (int)(2u << 16);        // data_size = 4 bytes
    g1[1] = (int)(DIMC << 16);      // tensor_dim0 = 768 (bits 79:48, low half)
    g1[2] = (int)(4u << 16);        // tensor_dim1 = 4   (bits 111:80, low half)
    g1[3] = (int)(DIMC << 16);      // tile_dim0   = 768 (bits 127:112)
    g1[4] = 4;                      // tile_dim1   = 4   (bits 143:128)
    g1[5] = DIMC;                   // tensor_dim0_stride = 768 (bits 207:160)
    g1[6] = 0;
    g1[7] = 0;
    v4i z4 = {0, 0, 0, 0};
#if __clang_major__ >= 23
    v8i z8 = {0, 0, 0, 0, 0, 0, 0, 0};
    __builtin_amdgcn_tensor_load_to_lds(g0, g1, z4, z4, z8, 0);
#else
    __builtin_amdgcn_tensor_load_to_lds(g0, g1, z4, z4, 0);
#endif
    __builtin_amdgcn_s_wait_tensorcnt(0);
  }
#else
  for (int t = threadIdx.x; t < 16 * DIMC; t += 128)
    Xf[t] = x[(size_t)row0 * DIMC + t];
  __syncthreads();
#endif

  // ---- LayerNorm: wave w normalizes rows 4w..4w+3 into LDS (f16) ----
  for (int i = 0; i < 4; ++i) {
    const int r = wave * 4 + i;
    float vals[DIMC / 32];
    float s = 0.f, ss = 0.f;
    #pragma unroll
    for (int j = 0; j < DIMC / 32; ++j) {
      float v = Xf[r * DIMC + lane + 32 * j];
      vals[j] = v; s += v; ss += v * v;
    }
    #pragma unroll
    for (int off = 16; off; off >>= 1) {
      s  += __shfl_xor(s,  off, 32);
      ss += __shfl_xor(ss, off, 32);
    }
    const float mu  = s * (1.f / DIMC);
    const float var = ss * (1.f / DIMC) - mu * mu;
    const float rs  = rsqrtf(var + 1e-5f);
    #pragma unroll
    for (int j = 0; j < DIMC / 32; ++j) {
      int c = lane + 32 * j;
      float xn = (vals[j] - mu) * rs * gamma[c] + beta[c];
      Ash[r * APAD + c] = (_Float16)xn;
    }
  }
  __syncthreads();

  // ---- GEMM: wave w owns N-tile w (cols 16w..16w+15), K=768 in 24 steps ----
  const int n    = wave * 16 + (lane & 15);
  const int koff = (lane & 16) ? 8 : 0;   // lanes 16-31 hold K {8..15,24..31}
  const int m    = lane & 15;
  v8f acc = {0.f, 0.f, 0.f, 0.f, 0.f, 0.f, 0.f, 0.f};
  for (int k0 = 0; k0 < DIMC; k0 += 32) {
    V16 A, Bv;
    A.h[0]  = *(const v8h*)&Ash[m * APAD + k0 + koff];
    A.h[1]  = *(const v8h*)&Ash[m * APAD + k0 + 16 + koff];
    Bv.h[0] = *(const v8h*)&WcT[n * DIMC + k0 + koff];
    Bv.h[1] = *(const v8h*)&WcT[n * DIMC + k0 + 16 + koff];
    acc = __builtin_amdgcn_wmma_f32_16x16x32_f16(false, A.v, false, Bv.v,
                                                 (short)0, acc, false, false);
  }

  if (n < 3*LD) {                      // cols 60..63 are padding
    const float bias  = bc[n];
    const int   mbase = (lane & 16) ? 8 : 0;
    #pragma unroll
    for (int r = 0; r < 8; ++r)
      qkv[(size_t)(row0 + mbase + r) * QKVP + n] = acc[r] + bias;
  }
}

// ---------------------------------------------------------------------------
// k2: local-window attention, one wave per query, online softmax.
//   o f32 [BN][32] (cols 20..31 zeroed for k3's padded-K WMMA)
// ---------------------------------------------------------------------------
__global__ __launch_bounds__(32) void k2_attn(const float* __restrict__ qkv,
                                              float* __restrict__ o) {
  __shared__ float pbuf[32];
  __shared__ int   kbuf[32];
  const int row = blockIdx.x;
  const int b   = row / NTOK;
  const int nq  = row - b * NTOK;
  const int qd  = nq / (HH * WW);
  const int rem = nq - qd * HH * WW;
  const int qh  = rem / WW;
  const int qw  = rem - qh * WW;

  // window (matches reference mask: d:[-1,1], h:[-3,2], w:[-3,2], clipped)
  const int dlo = max(qd - 1, 0), dhi = min(qd + 1, DD - 1);
  const int hlo = max(qh - 3, 0), hhi = min(qh + 2, HH - 1);
  const int wlo = max(qw - 3, 0), whi = min(qw + 2, WW - 1);
  const int nd = dhi - dlo + 1, nh = hhi - hlo + 1, nw = whi - wlo + 1;
  const int nk = nd * nh * nw;

  const int lane = threadIdx.x;
  const float* qrow = qkv + (size_t)row * QKVP;
  float q[LD];
  #pragma unroll
  for (int j = 0; j < LD; ++j) q[j] = qrow[j];

  const float scale = 0.03608439182435161f;  // 768^-0.5
  float M = -INFINITY, S = 0.f, acc = 0.f;   // acc: lane j<20 owns out[j]
  const int nchunk = (nk + 31) / 32;

  for (int c = 0; c < nchunk; ++c) {
    const int kk = c * 32 + lane;
    float sc = -INFINITY;
    int kn = 0;
    if (kk < nk) {
      int kd = kk / (nh * nw);
      int r2 = kk - kd * nh * nw;
      int kh = r2 / nw;
      int kw = r2 - kh * nw;
      kn = ((dlo + kd) * HH + (hlo + kh)) * WW + (wlo + kw);
      const float* krow = qkv + ((size_t)b * NTOK + kn) * QKVP + LD;
      float d = 0.f;
      #pragma unroll
      for (int j = 0; j < LD; ++j) d += q[j] * krow[j];
      sc = d * scale;
    }
    // wave max of this chunk
    float m = sc;
    #pragma unroll
    for (int off = 16; off; off >>= 1) m = fmaxf(m, __shfl_xor(m, off, 32));
    const float newM = fmaxf(M, m);
    const float p = (kk < nk) ? __expf(sc - newM) : 0.f;
    float psum = p;
    #pragma unroll
    for (int off = 16; off; off >>= 1) psum += __shfl_xor(psum, off, 32);
    const float f = __expf(M - newM);      // rescale factor (0 on first chunk)
    S = S * f + psum;

    pbuf[lane] = p;
    kbuf[lane] = kn;
    __syncthreads();
    const int cnt = min(nk - c * 32, 32);
    float a2 = 0.f;
    if (lane < LD) {
      for (int t = 0; t < cnt; ++t) {
        const float* vrow = qkv + ((size_t)b * NTOK + kbuf[t]) * QKVP + 2 * LD;
        a2 += pbuf[t] * vrow[lane];
      }
    }
    acc = acc * f + a2;
    __syncthreads();
    M = newM;
  }
  o[(size_t)row * OPAD + lane] = (lane < LD) ? (acc / S) : 0.f;
}

// ---------------------------------------------------------------------------
// k3: up-projection  out = o @ Wu + bu  via WMMA f16 (K padded 20->32).
//   grid = BN/16, 128 threads (4 waves x 12 N-tiles of 16 cols).
// ---------------------------------------------------------------------------
__global__ __launch_bounds__(128) void k3_up(const float* __restrict__ o,
                                             const _Float16* __restrict__ WuT,
                                             const float* __restrict__ bu,
                                             float* __restrict__ out) {
  __shared__ __attribute__((aligned(16))) _Float16 Ash[16 * 40];
  const int row0 = blockIdx.x * 16;
  const int lane = threadIdx.x & 31;
  const int wave = threadIdx.x >> 5;

  for (int t = threadIdx.x; t < 16 * OPAD; t += 128) {
    int r = t >> 5, cc = t & 31;
    Ash[r * 40 + cc] = (_Float16)o[(size_t)(row0 + r) * OPAD + cc];
  }
  __syncthreads();

  const int koff = (lane & 16) ? 8 : 0;
  const int m    = lane & 15;
  V16 A;
  A.h[0] = *(const v8h*)&Ash[m * 40 + koff];
  A.h[1] = *(const v8h*)&Ash[m * 40 + 16 + koff];

  const int mbase = (lane & 16) ? 8 : 0;
  for (int t = wave; t < DIMC / 16; t += 4) {
    const int c = t * 16 + (lane & 15);
    V16 Bv;
    Bv.h[0] = *(const v8h*)&WuT[c * OPAD + koff];
    Bv.h[1] = *(const v8h*)&WuT[c * OPAD + 16 + koff];
    v8f acc = {0.f, 0.f, 0.f, 0.f, 0.f, 0.f, 0.f, 0.f};
    acc = __builtin_amdgcn_wmma_f32_16x16x32_f16(false, A.v, false, Bv.v,
                                                 (short)0, acc, false, false);
    const float bias = bu[c];
    #pragma unroll
    for (int r = 0; r < 8; ++r)
      out[(size_t)(row0 + mbase + r) * DIMC + c] = acc[r] + bias;
  }
}

// ---------------------------------------------------------------------------
// launch
// ---------------------------------------------------------------------------
extern "C" void kernel_launch(void* const* d_in, const int* in_sizes, int n_in,
                              void* d_out, int out_size, void* d_ws, size_t ws_size,
                              hipStream_t stream) {
  const float* x     = (const float*)d_in[0];
  const float* gamma = (const float*)d_in[1];
  const float* beta  = (const float*)d_in[2];
  const float* Wd    = (const float*)d_in[3];
  const float* bd    = (const float*)d_in[4];
  const float* Wqkv  = (const float*)d_in[5];
  const float* Wu    = (const float*)d_in[6];
  const float* bu    = (const float*)d_in[7];
  // d_in[8] = mask: unused (window computed analytically, identical semantics)
  float* out = (float*)d_out;

  // workspace layout (256B-aligned slices)
  char* ws = (char*)d_ws;
  size_t off = 0;
  auto take = [&](size_t bytes) { char* p = ws + off; off = (off + bytes + 255) & ~(size_t)255; return p; };
  _Float16* WcT = (_Float16*)take((size_t)QKVP * DIMC * sizeof(_Float16)); // 96 KB
  _Float16* WuT = (_Float16*)take((size_t)DIMC * OPAD * sizeof(_Float16)); // 48 KB
  float*    bc  = (float*)   take((size_t)QKVP * sizeof(float));
  float*    qkv = (float*)   take((size_t)BN * QKVP * sizeof(float));      // 3.5 MB
  float*    ovb = (float*)   take((size_t)BN * OPAD * sizeof(float));      // 1.8 MB
  (void)ws_size; (void)in_sizes; (void)n_in; (void)out_size;

  k0_prep<<<QKVP + DIMC, 64, 0, stream>>>(Wd, bd, Wqkv, Wu, WcT, bc, WuT);
  k1_ln_qkv<<<BN / 16, 128, 0, stream>>>(x, gamma, beta, WcT, bc, qkv);
  k2_attn<<<BN, 32, 0, stream>>>(qkv, ovb);
  k3_up<<<BN / 16, 128, 0, stream>>>(ovb, WuT, bu, out);
}
